// TransformerEncoderLayer_82944408420460
// MI455X (gfx1250) — compile-verified, matrix operand order checked by reading, not run on hardware
//
#include <hip/hip_runtime.h>
#include <cstdint>
#include <cstddef>

// ---------------------------------------------------------------------------
// Transformer encoder layer for MI455X (gfx1250, wave32, WMMA).
// B=8, S=512, E=1024, H=16, D=64, FF=4096.  All matmuls in bf16 WMMA with
// f32 accumulate; softmax/LN in fp32.
// GEMMs: 128x128 block tile, 8 waves (each 32x64 = 8 WMMA tiles), K chunks of
// 32 staged in LDS via gfx1250 async global->LDS copies (ASYNCcnt) when the
// builtin exists, else a software-pipelined register path.
// ---------------------------------------------------------------------------

#define Bc 8
#define Sc 512
#define Ec 1024
#define Hc 16
#define Dc 64
#define FFc 4096
#define Mrows (Bc * Sc)   // 4096

typedef __attribute__((ext_vector_type(16))) __bf16 bf16x16;
typedef __attribute__((ext_vector_type(8)))  float  f32x8;
typedef int v4i __attribute__((vector_size(16)));   // matches builtin proto

#if __has_builtin(__builtin_amdgcn_global_load_async_to_lds_b128)
#define HAVE_ASYNC_LDS 1
#else
#define HAVE_ASYNC_LDS 0
#endif

#if HAVE_ASYNC_LDS
__device__ __forceinline__ void async_copy16(const void* g, void* l) {
  // proto (from compiler diagnostic): (v4i addrspace(1)*, v4i addrspace(3)*, imm, imm)
  __builtin_amdgcn_global_load_async_to_lds_b128(
      (__attribute__((address_space(1))) v4i*)g,
      (__attribute__((address_space(3))) v4i*)l, 0, 0);
}
#endif

template <int N>
__device__ __forceinline__ void wait_async() {
#if __has_builtin(__builtin_amdgcn_s_wait_asynccnt)
  __builtin_amdgcn_s_wait_asynccnt(N);
#else
  asm volatile("s_wait_asynccnt %0" ::"i"(N) : "memory");
#endif
}

// ---- CDNA5 bf16 16x16x32 WMMA fragment loader ------------------------------
// Per ISA 7.12.2 (16-bit A-matrix 16x32): lane = 16*half + idx holds row idx,
// K = {8*half .. +7} in VGPRs 0-3 and K = {16+8*half .. +7} in VGPRs 4-7.
// Both runs are 16-byte contiguous -> two uint4 reads per fragment.
// The B operand (32x16, K x N) is symmetric: lane holds column n = idx with
// the same K pattern, i.e. it reads row n of a row-major [N,K] source.
__device__ __forceinline__ bf16x16 load_frag(const __bf16* rowptr, int half) {
  union { uint4 u[2]; bf16x16 v; } f;
  const uint4* p = (const uint4*)rowptr;     // rowptr must be 16B aligned
  f.u[0] = p[half];
  f.u[1] = p[2 + half];
  return f.v;
}

__device__ __forceinline__ f32x8 wmma_bf16(bf16x16 a, bf16x16 b, f32x8 c) {
  return __builtin_amdgcn_wmma_f32_16x16x32_bf16(
      false, a, false, b, (short)0, c, false, false);
}

// ---------------------------------------------------------------------------
// fp32 -> bf16 elementwise conversion (weights)
// ---------------------------------------------------------------------------
__global__ void cvt_bf16_kernel(const float* __restrict__ in,
                                __bf16* __restrict__ out, int n) {
  int i = blockIdx.x * 256 + threadIdx.x;
  if (i < n) out[i] = (__bf16)in[i];
}

// ---------------------------------------------------------------------------
// LayerNorm over E=1024, one block per row, bf16 output
// ---------------------------------------------------------------------------
__global__ __launch_bounds__(256)
void ln_bf16_kernel(const float* __restrict__ x, const float* __restrict__ g,
                    const float* __restrict__ bta, __bf16* __restrict__ out) {
  __shared__ float sred[256];
  const int row = blockIdx.x, tid = threadIdx.x;
  const float* xr = x + (size_t)row * Ec;
  float v[4], s = 0.f;
#pragma unroll
  for (int i = 0; i < 4; ++i) { v[i] = xr[tid + i * 256]; s += v[i]; }
  sred[tid] = s; __syncthreads();
  for (int t = 128; t > 0; t >>= 1) {
    if (tid < t) sred[tid] += sred[tid + t];
    __syncthreads();
  }
  const float mu = sred[0] * (1.0f / Ec);
  __syncthreads();
  s = 0.f;
#pragma unroll
  for (int i = 0; i < 4; ++i) { float d = v[i] - mu; s += d * d; }
  sred[tid] = s; __syncthreads();
  for (int t = 128; t > 0; t >>= 1) {
    if (tid < t) sred[tid] += sred[tid + t];
    __syncthreads();
  }
  const float rsq = rsqrtf(sred[0] * (1.0f / Ec) + 1e-5f);
#pragma unroll
  for (int i = 0; i < 4; ++i) {
    int c = tid + i * 256;
    out[(size_t)row * Ec + c] = (__bf16)((v[i] - mu) * rsq * g[c] + bta[c]);
  }
}

// ---------------------------------------------------------------------------
// Generic bf16 WMMA GEMM:  out[M,N] = epilogue( A[M,K] @ W[N,K]^T + bias )
// Block tile 128(M) x 128(N), 8 waves; wave tile 32x64 (2x4 WMMA tiles).
// modes: 0 = bf16 out, scale, scatter to [B,H,S,D] (QKV)
//        1 = f32 out, += residual[m*N+n]
//        2 = bf16 out, exact GELU
// ---------------------------------------------------------------------------
#define LDA 40   // LDS row stride (bf16 elems), 80B = 5*16B aligned
__global__ __launch_bounds__(256)
void gemm_bf16_kernel(const __bf16* __restrict__ A, const __bf16* __restrict__ W,
                      const float* __restrict__ bias, const float* __restrict__ res,
                      void* __restrict__ outp, int M, int N, int K,
                      int mode, float scale) {
  constexpr int NBUF = HAVE_ASYNC_LDS ? 2 : 1;
  __shared__ __bf16 As[NBUF][128 * LDA];
  __shared__ __bf16 Bs[NBUF][128 * LDA];
  const int tid = threadIdx.x;
  const int bm = blockIdx.y * 128, bn = blockIdx.x * 128;
  const int wave = tid >> 5, lane = tid & 31;
  const int half = lane >> 4, idx = lane & 15;
  const int wm = (wave & 3) * 32, wn = (wave >> 2) * 64;
  // staging coordinates: 512 uint4 per matrix, 2 per thread per matrix
  const int r0 = tid >> 2, c0 = tid & 3;            // lin = tid
  const int r1 = (tid + 256) >> 2, c1 = tid & 3;    // lin = tid + 256

  f32x8 acc[2][4] = {};
  const int nk = K >> 5;

#if HAVE_ASYNC_LDS
  // ---- async double-buffered pipeline (GLOBAL_LOAD_ASYNC_TO_LDS_B128) ----
  auto issue = [&](int kb, int buf) {
    async_copy16(A + (size_t)(bm + r0) * K + kb + c0 * 8, &As[buf][r0 * LDA + c0 * 8]);
    async_copy16(A + (size_t)(bm + r1) * K + kb + c1 * 8, &As[buf][r1 * LDA + c1 * 8]);
    async_copy16(W + (size_t)(bn + r0) * K + kb + c0 * 8, &Bs[buf][r0 * LDA + c0 * 8]);
    async_copy16(W + (size_t)(bn + r1) * K + kb + c1 * 8, &Bs[buf][r1 * LDA + c1 * 8]);
  };
  issue(0, 0);
  for (int kk = 0; kk < nk; ++kk) {
    const int cur = kk & 1;
    if (kk + 1 < nk) { issue((kk + 1) << 5, cur ^ 1); wait_async<4>(); }
    else             { wait_async<0>(); }
    __syncthreads();
    const __bf16* Ab = As[cur];
    const __bf16* Bb = Bs[cur];
    bf16x16 a0 = load_frag(Ab + (wm + idx) * LDA, half);
    bf16x16 a1 = load_frag(Ab + (wm + 16 + idx) * LDA, half);
#pragma unroll
    for (int t = 0; t < 4; ++t) {
      bf16x16 bt = load_frag(Bb + (wn + t * 16 + idx) * LDA, half);
      acc[0][t] = wmma_bf16(a0, bt, acc[0][t]);
      acc[1][t] = wmma_bf16(a1, bt, acc[1][t]);
    }
    __syncthreads();
  }
#else
  // ---- software-pipelined sync path: next chunk loads overlap WMMA -------
  uint4 ra0, ra1, rb0, rb1;
  auto gload = [&](int kb) {
    ra0 = *(const uint4*)(A + (size_t)(bm + r0) * K + kb + c0 * 8);
    ra1 = *(const uint4*)(A + (size_t)(bm + r1) * K + kb + c1 * 8);
    rb0 = *(const uint4*)(W + (size_t)(bn + r0) * K + kb + c0 * 8);
    rb1 = *(const uint4*)(W + (size_t)(bn + r1) * K + kb + c1 * 8);
  };
  gload(0);
  for (int kk = 0; kk < nk; ++kk) {
    *(uint4*)(&As[0][r0 * LDA + c0 * 8]) = ra0;
    *(uint4*)(&As[0][r1 * LDA + c1 * 8]) = ra1;
    *(uint4*)(&Bs[0][r0 * LDA + c0 * 8]) = rb0;
    *(uint4*)(&Bs[0][r1 * LDA + c1 * 8]) = rb1;
    __syncthreads();
    if (kk + 1 < nk) gload((kk + 1) << 5);   // overlaps the WMMAs below
    bf16x16 a0 = load_frag(&As[0][(wm + idx) * LDA], half);
    bf16x16 a1 = load_frag(&As[0][(wm + 16 + idx) * LDA], half);
#pragma unroll
    for (int t = 0; t < 4; ++t) {
      bf16x16 bt = load_frag(&Bs[0][(wn + t * 16 + idx) * LDA], half);
      acc[0][t] = wmma_bf16(a0, bt, acc[0][t]);
      acc[1][t] = wmma_bf16(a1, bt, acc[1][t]);
    }
    __syncthreads();
  }
#endif

  // Epilogue. C layout: lane(16*half+idx), VGPR r -> row r+8*half, col idx.
#pragma unroll
  for (int im = 0; im < 2; ++im)
#pragma unroll
    for (int in_ = 0; in_ < 4; ++in_)
#pragma unroll
      for (int r = 0; r < 8; ++r) {
        const int m = bm + wm + im * 16 + r + half * 8;
        const int n = bn + wn + in_ * 16 + idx;
        float v = acc[im][in_][r] + bias[n];
        if (mode == 0) {           // QKV: scale + permute to [B,H,S,D] bf16
          v *= scale;
          int bb = m >> 9, s = m & 511, hh = n >> 6, d = n & 63;
          ((__bf16*)outp)[(((size_t)bb * Hc + hh) * Sc + s) * Dc + d] = (__bf16)v;
        } else if (mode == 1) {    // residual add, f32 out
          ((float*)outp)[(size_t)m * N + n] = v + res[(size_t)m * N + n];
        } else {                   // exact GELU, bf16 out
          float ge = 0.5f * v * (1.0f + erff(v * 0.70710678118654752f));
          ((__bf16*)outp)[(size_t)m * N + n] = (__bf16)ge;
        }
      }
}

// ---------------------------------------------------------------------------
// Attention: one WG (8 waves) per (b, h, 16-query tile).
//   logits(16x512) = Qs @ K^T  (WMMA; K rows are read directly from global —
//   each lane's B-fragment reads are the contiguous row K[n, k..k+7])
//   fp32 softmax with mask, + post-softmax adjoin, bf16 P in LDS
//   xv(16x64) = P @ V  (V chunk transposed into LDS so B-frag reads are
//   contiguous), written back merged as [B,S,E] bf16.
// Static LDS = 62,016 B < 64 KB.
// ---------------------------------------------------------------------------
__global__ __launch_bounds__(256)
void attn_kernel(const __bf16* __restrict__ Q, const __bf16* __restrict__ Kt,
                 const __bf16* __restrict__ V, const float* __restrict__ mask,
                 const float* __restrict__ adjoin, __bf16* __restrict__ xv) {
  __shared__ __bf16 Qs[16 * 72];
  __shared__ float  Lg[16 * 512];
  __shared__ __bf16 Ps[16 * 520];
  __shared__ __bf16 Vt[64 * 72];
  __shared__ float  red[16 * 17];

  const int tid = threadIdx.x;
  const int qt = blockIdx.x, h = blockIdx.y, b = blockIdx.z;
  const int q0 = qt * 16;
  const size_t bh = ((size_t)b * Hc + h) * Sc;
  const int wave = tid >> 5, lane = tid & 31;
  const int half = lane >> 4, idx = lane & 15;

  {  // load Q tile (pre-scaled by 1/sqrt(D) in the QKV epilogue)
    int row = tid >> 4, ch = tid & 15;
    uint2 v = *(const uint2*)(Q + (bh + q0 + row) * Dc + ch * 4);
    *(uint2*)(Qs + row * 72 + ch * 4) = v;
  }
  __syncthreads();

  {  // logits = Q K^T : 32 col-tiles, 4 per wave
    bf16x16 a0 = load_frag(Qs + idx * 72, half);
    bf16x16 a1 = load_frag(Qs + idx * 72 + 32, half);
#pragma unroll
    for (int i = 0; i < 4; ++i) {
      const int ct = wave * 4 + i;
      const __bf16* krow = Kt + (bh + ct * 16 + idx) * Dc;
      bf16x16 b0 = load_frag(krow, half);
      bf16x16 b1 = load_frag(krow + 32, half);
      f32x8 c = {};
      c = wmma_bf16(a0, b0, c);
      c = wmma_bf16(a1, b1, c);
#pragma unroll
      for (int r = 0; r < 8; ++r)
        Lg[(r + half * 8) * 512 + ct * 16 + idx] = c[r];
    }
  }
  __syncthreads();

  {  // mask + softmax + adjoin (16 rows x 16 threads, 32 cols/thread)
    const int row = tid >> 4, l16 = tid & 15;
    const int sq = q0 + row;
    const float* mrow = mask + ((size_t)b * Sc + sq) * Sc;
    const float* arow = adjoin + ((size_t)b * Sc + sq) * Sc;
    float lv[32], mx = -3.0e38f;
#pragma unroll
    for (int j = 0; j < 32; ++j) {
      int c = l16 + j * 16;
      float v = Lg[row * 512 + c] + mrow[c] * -1e9f;
      lv[j] = v; mx = fmaxf(mx, v);
    }
    red[row * 17 + l16] = mx; __syncthreads();
    for (int s = 8; s > 0; s >>= 1) {
      if (l16 < s) red[row * 17 + l16] =
          fmaxf(red[row * 17 + l16], red[row * 17 + l16 + s]);
      __syncthreads();
    }
    mx = red[row * 17]; __syncthreads();
    float sum = 0.f;
#pragma unroll
    for (int j = 0; j < 32; ++j) { lv[j] = __expf(lv[j] - mx); sum += lv[j]; }
    red[row * 17 + l16] = sum; __syncthreads();
    for (int s = 8; s > 0; s >>= 1) {
      if (l16 < s) red[row * 17 + l16] += red[row * 17 + l16 + s];
      __syncthreads();
    }
    const float inv = 1.0f / red[row * 17];
#pragma unroll
    for (int j = 0; j < 32; ++j) {
      int c = l16 + j * 16;
      Ps[row * 520 + c] = (__bf16)(lv[j] * inv + arow[c]);
    }
  }
  __syncthreads();

  // xv = P @ V, streaming 64-key V chunks through LDS (transposed)
  f32x8 o = {};
  const int dtile = wave & 3;
  for (int kc = 0; kc < 8; ++kc) {
    {
      int key = tid >> 2, seg = tid & 3, d0 = seg * 16;
      const __bf16* vrow = V + (bh + kc * 64 + key) * Dc + d0;
      union { uint4 u; __bf16 e[8]; } w0, w1;
      w0.u = *(const uint4*)(vrow);
      w1.u = *(const uint4*)(vrow + 8);
#pragma unroll
      for (int j = 0; j < 8; ++j) Vt[(d0 + j) * 72 + key] = w0.e[j];
#pragma unroll
      for (int j = 0; j < 8; ++j) Vt[(d0 + 8 + j) * 72 + key] = w1.e[j];
    }
    __syncthreads();
    if (wave < 4) {
      bf16x16 a0 = load_frag(Ps + idx * 520 + kc * 64, half);
      bf16x16 a1 = load_frag(Ps + idx * 520 + kc * 64 + 32, half);
      bf16x16 b0 = load_frag(Vt + (dtile * 16 + idx) * 72, half);
      bf16x16 b1 = load_frag(Vt + (dtile * 16 + idx) * 72 + 32, half);
      o = wmma_bf16(a0, b0, o);
      o = wmma_bf16(a1, b1, o);
    }
    __syncthreads();
  }
  if (wave < 4) {
#pragma unroll
    for (int r = 0; r < 8; ++r) {
      int sq = q0 + r + half * 8;
      int d = dtile * 16 + idx;
      xv[((size_t)b * Sc + sq) * Ec + h * Dc + d] = (__bf16)o[r];
    }
  }
}

// ---------------------------------------------------------------------------
// Host-side orchestration
// ---------------------------------------------------------------------------
extern "C" void kernel_launch(void* const* d_in, const int* in_sizes, int n_in,
                              void* d_out, int out_size, void* d_ws, size_t ws_size,
                              hipStream_t stream) {
  const float* x      = (const float*)d_in[0];
  const float* mask   = (const float*)d_in[1];
  const float* adjoin = (const float*)d_in[2];
  const float* Wq = (const float*)d_in[3];  const float* bq  = (const float*)d_in[4];
  const float* Wk = (const float*)d_in[5];  const float* bk  = (const float*)d_in[6];
  const float* Wv = (const float*)d_in[7];  const float* bv  = (const float*)d_in[8];
  const float* Wo = (const float*)d_in[9];  const float* bo  = (const float*)d_in[10];
  const float* g1 = (const float*)d_in[11]; const float* b1  = (const float*)d_in[12];
  const float* g2 = (const float*)d_in[13]; const float* b2  = (const float*)d_in[14];
  const float* W1 = (const float*)d_in[15]; const float* bb1 = (const float*)d_in[16];
  const float* W2 = (const float*)d_in[17]; const float* bb2 = (const float*)d_in[18];

  char* ws = (char*)d_ws;
  size_t off = 0;
  auto alloc = [&](size_t bytes) {
    void* p = ws + off;
    off += (bytes + 255) & ~(size_t)255;
    return p;
  };
  const size_t EE = (size_t)Ec * Ec, FE = (size_t)FFc * Ec;
  __bf16* Wq_b = (__bf16*)alloc(EE * 2);
  __bf16* Wk_b = (__bf16*)alloc(EE * 2);
  __bf16* Wv_b = (__bf16*)alloc(EE * 2);
  __bf16* Wo_b = (__bf16*)alloc(EE * 2);
  __bf16* W1_b = (__bf16*)alloc(FE * 2);
  __bf16* W2_b = (__bf16*)alloc(FE * 2);
  __bf16* nx_b = (__bf16*)alloc((size_t)Mrows * Ec * 2);   // reused for nx2
  __bf16* Q_b  = (__bf16*)alloc((size_t)Mrows * Ec * 2);
  __bf16* K_b  = (__bf16*)alloc((size_t)Mrows * Ec * 2);
  __bf16* V_b  = (__bf16*)alloc((size_t)Mrows * Ec * 2);
  __bf16* xv_b = (__bf16*)alloc((size_t)Mrows * Ec * 2);
  float*  x1_f = (float*) alloc((size_t)Mrows * Ec * 4);
  __bf16* h_b  = (__bf16*)alloc((size_t)Mrows * FFc * 2);
  (void)ws_size; (void)in_sizes; (void)n_in; (void)out_size;

  const dim3 blk(256);
  cvt_bf16_kernel<<<(int)((EE + 255) / 256), blk, 0, stream>>>(Wq, Wq_b, (int)EE);
  cvt_bf16_kernel<<<(int)((EE + 255) / 256), blk, 0, stream>>>(Wk, Wk_b, (int)EE);
  cvt_bf16_kernel<<<(int)((EE + 255) / 256), blk, 0, stream>>>(Wv, Wv_b, (int)EE);
  cvt_bf16_kernel<<<(int)((EE + 255) / 256), blk, 0, stream>>>(Wo, Wo_b, (int)EE);
  cvt_bf16_kernel<<<(int)((FE + 255) / 256), blk, 0, stream>>>(W1, W1_b, (int)FE);
  cvt_bf16_kernel<<<(int)((FE + 255) / 256), blk, 0, stream>>>(W2, W2_b, (int)FE);

  ln_bf16_kernel<<<Mrows, blk, 0, stream>>>(x, g1, b1, nx_b);

  const dim3 gE(Ec / 128, Mrows / 128);
  gemm_bf16_kernel<<<gE, blk, 0, stream>>>(nx_b, Wq_b, bq, nullptr, Q_b,
                                           Mrows, Ec, Ec, 0, 0.125f);
  gemm_bf16_kernel<<<gE, blk, 0, stream>>>(nx_b, Wk_b, bk, nullptr, K_b,
                                           Mrows, Ec, Ec, 0, 1.0f);
  gemm_bf16_kernel<<<gE, blk, 0, stream>>>(nx_b, Wv_b, bv, nullptr, V_b,
                                           Mrows, Ec, Ec, 0, 1.0f);

  attn_kernel<<<dim3(Sc / 16, Hc, Bc), blk, 0, stream>>>(Q_b, K_b, V_b, mask,
                                                         adjoin, xv_b);

  gemm_bf16_kernel<<<gE, blk, 0, stream>>>(xv_b, Wo_b, bo, x, x1_f,
                                           Mrows, Ec, Ec, 1, 1.0f);
  ln_bf16_kernel<<<Mrows, blk, 0, stream>>>(x1_f, g2, b2, nx_b);
  gemm_bf16_kernel<<<dim3(FFc / 128, Mrows / 128), blk, 0, stream>>>(
      nx_b, W1_b, bb1, nullptr, h_b, Mrows, FFc, Ec, 2, 1.0f);
  gemm_bf16_kernel<<<gE, blk, 0, stream>>>(h_b, W2_b, bb2, x1_f, (float*)d_out,
                                           Mrows, Ec, FFc, 1, 1.0f);
}
